// CoordinateWiseLSTMOptimizer_23081154249180
// MI455X (gfx1250) — compile-verified, matrix operand order checked by reading, not run on hardware
//
#include <hip/hip_runtime.h>

typedef __attribute__((ext_vector_type(16))) _Float16 v16h;
typedef __attribute__((ext_vector_type(8)))  float    v8f;

#define HID 20           // LSTM hidden size
#define TPW 8            // 16-coordinate tiles per wave
#define THREADS 256
#define WAVES (THREADS / 32)

__device__ __forceinline__ float fast_sigmoid(float x) {
    float t = __builtin_amdgcn_exp2f(-1.44269504088896f * x);
    return __builtin_amdgcn_rcpf(1.0f + t);
}
__device__ __forceinline__ float fast_tanh(float x) {
    float t = __builtin_amdgcn_exp2f(-2.88539008177793f * x);
    return 2.0f * __builtin_amdgcn_rcpf(1.0f + t) - 1.0f;
}

__global__ __launch_bounds__(THREADS)
void lstm_opt_wmma_kernel(const float* __restrict__ params,
                          const float* __restrict__ grads,
                          const float* __restrict__ h0,
                          const float* __restrict__ c0,
                          const float* __restrict__ W_ih,   // [80,2]
                          const float* __restrict__ W_hh,   // [80,20]
                          const float* __restrict__ b_ih,   // [80]
                          const float* __restrict__ b_hh,   // [80]
                          const float* __restrict__ W_out,  // [1,20]
                          const float* __restrict__ b_out,  // [1]
                          float* __restrict__ out, int n) {
    const int lane = threadIdx.x & 31;
    const int wave = threadIdx.x >> 5;
    const int gw   = blockIdx.x * WAVES + wave;

    const int  nloc = lane & 15;      // column within a 16-wide tile
    const bool hi   = lane >= 16;     // upper half-wave (K-halves / M rows 8..15)

    // ---- One-time per-wave setup: B tiles (f16), biases, output weights ----
    // Output-column permutation: tiles 0..3 = gate i/f/g/o for hidden 0..15;
    // tile 4 packs 4 gates x hidden 16..19 (col = gate*4 + (h-16)).
    v16h  Bm[5];
    float bias[5];
#pragma unroll
    for (int t = 0; t < 5; ++t) {
        int row = (t < 4) ? (t * HID + nloc)
                          : ((nloc >> 2) * HID + 16 + (nloc & 3)); // PyTorch i,f,g,o order
        bias[t] = b_ih[row] + b_hh[row];
#pragma unroll
        for (int j = 0; j < 16; ++j) {
            // 16-bit B operand K mapping (mirrors A layout):
            // lanes 0-15:  j<8 -> K=j,   j>=8 -> K=j+8  (16..23)
            // lanes 16-31: j<8 -> K=j+8, j>=8 -> K=j+16 (24..31)
            int k = hi ? ((j < 8) ? (j + 8) : (j + 16))
                       : ((j < 8) ? j : (j + 8));
            float w;
            if      (k == 0)      w = W_ih[row * 2 + 0];          // grad weight
            else if (k == 1)      w = W_ih[row * 2 + 1];          // param weight
            else if (k < 2 + HID) w = W_hh[row * HID + (k - 2)];  // h0 weights
            else                  w = 0.0f;                       // K pad 22..31
            Bm[t][j] = (_Float16)w;
        }
    }
    const float w_out_main = W_out[nloc];            // hidden 0..15 (lane-aligned)
    const float w_out4     = W_out[16 + (lane & 3)]; // hidden 16..19
    const float bo         = b_out[0];
    const int   hh         = lane & 3;               // tile4 hidden offset
    const int   srcb       = lane & 16;              // shuffle base for half-wave
    // Tile-4 unified activation: av = s * rcp(1 + exp2(mul*x)) + c
    //   sigmoid lanes: mul=-log2e,  s=1, c=0 ; tanh lanes (grp==2): mul=-2log2e, s=2, c=-1
    const bool  isg4  = ((nloc >> 2) == 2);
    const float mul4  = isg4 ? -2.88539008177793f : -1.44269504088896f;
    const float s4    = isg4 ? 2.0f : 1.0f;
    const float c4    = isg4 ? -1.0f : 0.0f;

    // Persistent A fragment; zero-pad elements written once (WMMA never clobbers A).
    v16h a;
#pragma unroll
    for (int j = 0; j < 16; ++j) a[j] = (_Float16)0.0f;

    // ---- Main loop: one 16-coordinate tile per iteration ----
    for (int tt = 0; tt < TPW; ++tt) {
        int base = (gw * TPW + tt) * 16;
        if (base + 16 > n) break;                    // wave-uniform

        // Build A (16x32 f16): K=0 grad, K=1 param, K=2..21 h0, K>=22 zero.
        {
            int m = base + nloc;
            const float* h0r = h0 + (size_t)m * HID;   // 80B rows -> 16B aligned
            if (!hi) {
                a[0] = (_Float16)grads[m];
                a[1] = (_Float16)params[m];
                float4 p0 = *(const float4*)(h0r);        // h0[0..3]   (K=2..5)
                float2 p1 = *(const float2*)(h0r + 4);    // h0[4..5]   (K=6..7)
                float2 p2 = *(const float2*)(h0r + 14);   // h0[14..15] (K=16..17)
                float4 p3 = *(const float4*)(h0r + 16);   // h0[16..19] (K=18..21)
                a[2]  = (_Float16)p0.x; a[3]  = (_Float16)p0.y;
                a[4]  = (_Float16)p0.z; a[5]  = (_Float16)p0.w;
                a[6]  = (_Float16)p1.x; a[7]  = (_Float16)p1.y;
                a[8]  = (_Float16)p2.x; a[9]  = (_Float16)p2.y;
                a[10] = (_Float16)p3.x; a[11] = (_Float16)p3.y;
                a[12] = (_Float16)p3.z; a[13] = (_Float16)p3.w;
                // a[14], a[15] stay zero (K=22,23)
            } else {
                float2 q0 = *(const float2*)(h0r + 6);    // h0[6..7]   (K=8..9)
                float4 q1 = *(const float4*)(h0r + 8);    // h0[8..11]  (K=10..13)
                float2 q2 = *(const float2*)(h0r + 12);   // h0[12..13] (K=14..15)
                a[0] = (_Float16)q0.x; a[1] = (_Float16)q0.y;
                a[2] = (_Float16)q1.x; a[3] = (_Float16)q1.y;
                a[4] = (_Float16)q1.z; a[5] = (_Float16)q1.w;
                a[6] = (_Float16)q2.x; a[7] = (_Float16)q2.y;
                // a[8..15] stay zero (K=24..31)
            }
        }

        // 5 WMMAs, bias pre-loaded into C.
        v8f acc[5];
#pragma unroll
        for (int t = 0; t < 5; ++t) {
            v8f c;
#pragma unroll
            for (int r = 0; r < 8; ++r) c[r] = bias[t];
            acc[t] = __builtin_amdgcn_wmma_f32_16x16x32_f16(
                false, a, false, Bm[t], (short)0, c, false, false);
        }

        // Row-base pointers once per tile -> c0 loads become imm-offset B32s.
        const int mrow = base + (hi ? 8 : 0);        // coordinate row for vgpr index r
        const float* c0r  = c0 + (size_t)mrow * HID + nloc;     // main, col = nloc
        const float* c0r4 = c0 + (size_t)mrow * HID + 16 + hh;  // tile4, col = 16+hh

        // Elementwise gates for hidden 0..15 (fully lane-aligned in C/D layout).
        float upd[8];
#pragma unroll
        for (int r = 0; r < 8; ++r) {
            float iv  = fast_sigmoid(acc[0][r]);
            float fv  = fast_sigmoid(acc[1][r]);
            float gv  = fast_tanh   (acc[2][r]);
            float ov  = fast_sigmoid(acc[3][r]);
            float c0v = c0r[r * HID];
            float c1  = fv * c0v + iv * gv;
            upd[r]    = ov * fast_tanh(c1) * w_out_main;
        }
        // Reduce over hidden 0..15 (lanes within each 16-lane half).
#pragma unroll
        for (int r = 0; r < 8; ++r) {
            upd[r] += __shfl_xor(upd[r], 1, 32);
            upd[r] += __shfl_xor(upd[r], 2, 32);
            upd[r] += __shfl_xor(upd[r], 4, 32);
            upd[r] += __shfl_xor(upd[r], 8, 32);
        }

        // Tile 4: hidden 16..19. One-formula activation, realign with shuffles.
#pragma unroll
        for (int r = 0; r < 8; ++r) {
            float v  = acc[4][r];
            float av = s4 * __builtin_amdgcn_rcpf(1.0f + __builtin_amdgcn_exp2f(mul4 * v)) + c4;
            float iv = __shfl(av, srcb + 0  + hh, 32);
            float fv = __shfl(av, srcb + 4  + hh, 32);
            float gv = __shfl(av, srcb + 8  + hh, 32);
            float ov = __shfl(av, srcb + 12 + hh, 32);
            float c0v = c0r4[r * HID];
            float c1  = fv * c0v + iv * gv;
            float e   = ov * fast_tanh(c1) * w_out4;
            e += __shfl_xor(e, 1, 32);               // sum over 4 hidden units
            e += __shfl_xor(e, 2, 32);
            upd[r] += e;                             // identical across the 16-half
        }

        // Store: lane 0 -> rows base..base+7, lane 16 -> rows base+8..base+15.
        if (lane == 0 || lane == 16) {
            float* o = out + base + (hi ? 8 : 0);    // 16-float aligned -> B128 ok
            float4 v0 = make_float4(upd[0] + bo, upd[1] + bo, upd[2] + bo, upd[3] + bo);
            float4 v1 = make_float4(upd[4] + bo, upd[5] + bo, upd[6] + bo, upd[7] + bo);
            *(float4*)(o)     = v0;
            *(float4*)(o + 4) = v1;
        }
    }

    // ---- Scalar tail for n not a multiple of 16 (n=2M is, but stay safe) ----
    int rem = n & 15;
    if (rem && blockIdx.x == 0 && (int)threadIdx.x < rem) {
        int m = (n & ~15) + threadIdx.x;
        float u = bo;
        float xg = grads[m], xp = params[m];
        for (int h = 0; h < HID; ++h) {
            float g4[4];
#pragma unroll
            for (int t = 0; t < 4; ++t) {
                int row = t * HID + h;
                float acc = b_ih[row] + b_hh[row] + xg * W_ih[row * 2] + xp * W_ih[row * 2 + 1];
                for (int k = 0; k < HID; ++k)
                    acc += h0[(size_t)m * HID + k] * W_hh[row * HID + k];
                g4[t] = acc;
            }
            float iv = fast_sigmoid(g4[0]);
            float fv = fast_sigmoid(g4[1]);
            float gv = fast_tanh(g4[2]);
            float ov = fast_sigmoid(g4[3]);
            float c1 = fv * c0[(size_t)m * HID + h] + iv * gv;
            u += ov * fast_tanh(c1) * W_out[h];
        }
        out[m] = u;
    }
}

extern "C" void kernel_launch(void* const* d_in, const int* in_sizes, int n_in,
                              void* d_out, int out_size, void* d_ws, size_t ws_size,
                              hipStream_t stream) {
    const float* params = (const float*)d_in[0];
    const float* grads  = (const float*)d_in[1];
    const float* h0     = (const float*)d_in[2];
    const float* c0     = (const float*)d_in[3];
    const float* W_ih   = (const float*)d_in[4];
    const float* W_hh   = (const float*)d_in[5];
    const float* b_ih   = (const float*)d_in[6];
    const float* b_hh   = (const float*)d_in[7];
    const float* W_out  = (const float*)d_in[8];
    const float* b_out  = (const float*)d_in[9];
    float* out = (float*)d_out;
    int n = in_sizes[0];

    int coords_per_block = WAVES * TPW * 16;     // 1024
    int blocks = (n + coords_per_block - 1) / coords_per_block;
    if (blocks < 1) blocks = 1;
    lstm_opt_wmma_kernel<<<blocks, THREADS, 0, stream>>>(
        params, grads, h0, c0, W_ih, W_hh, b_ih, b_hh, W_out, b_out, out, n);
}
